// LSTM_19859928777266
// MI455X (gfx1250) — matrix operand order repaired and checked by reading, not compiled, not run on hardware
//
#include <hip/hip_runtime.h>

// ---------------------------------------------------------------------------
// 2-layer LSTM for MI455X (gfx1250, wave32, WMMA).
// B=128, NUMS=128, SEQ=256, H=1024, PRED=96.
//  - bf16 WMMA 16x16x32, fp32 accumulation (8x more K/instr than f32 WMMA)
//  - per-WG weight slice (147KB/262KB) resident in 320KB LDS for the whole
//    layer: B operands come from ds_load_b128, zero per-step weight traffic
//  - chunk-major LDS layout: all 8 ds_loads of a chunk are 16-bit immediate
//    offsets off one incrementing base VGPR (no per-chunk VALU addressing)
//  - persistent kernel, monotonic-counter grid barrier per timestep
//  - cell state c lives in VGPRs for all 512 steps; 4 gates fused in regs
//  - straight-line sigmoid/tanh via v_exp + fast divide (no libm branches)
// ---------------------------------------------------------------------------

#define BATCH 128
#define NUMS  128
#define SEQL  256
#define HID   1024
#define PRED  96
#define NWG   64          // HID/16 workgroups in persistent kernel

typedef __attribute__((ext_vector_type(16))) __bf16 v16bf;
typedef __attribute__((ext_vector_type(8)))  float  v8f;

union Frag { uint4 q[2]; v16bf v; };

// A fragment (16x32 bf16, M x K), row-major source with row stride `stride`.
// ISA layout: lanes0-15 -> M=lane, k in {kb..kb+7, kb+16..kb+23}, kb=(lane>>4)*8
__device__ __forceinline__ v16bf load_a_frag(const __bf16* __restrict__ src,
                                             int stride, int lane) {
  int r  = lane & 15;
  int kb = (lane >> 4) * 8;
  const __bf16* p = src + (size_t)r * stride + kb;
  Frag f;
  f.q[0] = *reinterpret_cast<const uint4*>(p);
  f.q[1] = *reinterpret_cast<const uint4*>(p + 16);
  return f.v;
}

// B fragment (32x16 bf16, K x N) from N-major weights Wt[N][K]:
// lanes0-15 -> N=lane, K={0..7,16..23}; lanes16-31 -> N=lane-16, K={8..15,24..31}
__device__ __forceinline__ v16bf load_b_frag(const __bf16* __restrict__ wrow0,
                                             size_t stride, int lane) {
  int n  = lane & 15;
  int k0 = (lane >> 4) * 8;
  const __bf16* p = wrow0 + (size_t)n * stride + k0;
  Frag f;
  f.q[0] = *reinterpret_cast<const uint4*>(p);
  f.q[1] = *reinterpret_cast<const uint4*>(p + 16);
  return f.v;
}

// Straight-line activations: one v_exp_f32 + fast divide, no divergence.
__device__ __forceinline__ float fast_sigmoid(float x) {
  float e = __expf(-x);
  return __fdividef(1.0f, 1.0f + e);
}
__device__ __forceinline__ float fast_tanh(float x) {
  float ax = __builtin_fabsf(x);
  float e  = __expf(-2.0f * ax);
  float t  = __fdividef(1.0f - e, 1.0f + e);
  return __builtin_copysignf(t, x);
}

// Monotonic-counter grid barrier (64 WGs co-resident; counter zeroed per call).
__device__ __forceinline__ void grid_barrier(int* bar, int target) {
  __threadfence();
  __syncthreads();
  if (threadIdx.x == 0) {
    atomicAdd(bar, 1);
    while (__hip_atomic_load(bar, __ATOMIC_RELAXED, __HIP_MEMORY_SCOPE_AGENT)
           < target) {
      __builtin_amdgcn_s_sleep(1);
    }
  }
  __syncthreads();
  __threadfence();
}

// ---------------------------------------------------------------------------
// Prep kernels
// ---------------------------------------------------------------------------

// dst[n*K + k] = (bf16) src[k*N + n]   (transpose KxN fp32 -> NxK bf16)
__global__ void transpose_bf16(const float* __restrict__ src,
                               __bf16* __restrict__ dst, int K, int N) {
  int idx = blockIdx.x * blockDim.x + threadIdx.x;
  if (idx >= K * N) return;
  int n = idx / K;
  int k = idx - n * K;
  dst[idx] = (__bf16)src[(size_t)k * N + n];
}

// xT[t][b][d] = (bf16) x[b][d][t]
__global__ void xpose_bf16(const float* __restrict__ x,
                           __bf16* __restrict__ xT) {
  int idx = blockIdx.x * blockDim.x + threadIdx.x;
  if (idx >= SEQL * BATCH * NUMS) return;
  int t = idx / (BATCH * NUMS);
  int r = idx - t * (BATCH * NUMS);
  int b = r / NUMS;
  int d = r - b * NUMS;
  xT[idx] = (__bf16)x[((size_t)b * NUMS + d) * SEQL + t];
}

// zero h0 slot and barrier counter
__global__ void init_state(__bf16* __restrict__ hs0, int* __restrict__ bar) {
  int i = blockIdx.x * blockDim.x + threadIdx.x;
  if (i == 0) *bar = 0;
  if (i < BATCH * HID) hs0[i] = (__bf16)0.0f;
}

// ---------------------------------------------------------------------------
// Persistent 2-layer LSTM recurrence.
// Grid: 64 WGs (one per 16 hidden cols) x 256 threads (8 waves = M tiles).
// Dynamic LDS: (K/32) chunks * 4KB, chunk-major (max 256KB for layer 1).
// LDS offset of chunk c, gate g, lane l, half h:
//   c*4096 + g*1024 + ((l*32 + h*16) ^ ((l>>3)&1)*16)   [bank-conflict-free]
// ---------------------------------------------------------------------------
extern __shared__ char smem[];

__global__ void __launch_bounds__(256, 1)
lstm_persistent(const float* __restrict__ b0, const float* __restrict__ b1,
                const __bf16* __restrict__ Wt0,   // [4H][K0] bf16, K0=1152
                const __bf16* __restrict__ Wt1,   // [4H][K1] bf16, K1=2048
                const __bf16* __restrict__ xT,    // [T][B][NUMS] bf16
                __bf16* __restrict__ hs0,         // [T+1][B][H] bf16 (slot0=0)
                __bf16* __restrict__ h1,          // [2][B][H] bf16 ping-pong
                float* __restrict__ out_h, float* __restrict__ out_c,
                int* __restrict__ bar) {
  const int lane  = threadIdx.x & 31;
  const int wave  = threadIdx.x >> 5;   // M tile: rows [wave*16, wave*16+16)
  const int mbase = wave * 16;
  const int n0    = blockIdx.x * 16;    // hidden column slice
  const int nlane = n0 + (lane & 15);
  const size_t BH = (size_t)BATCH * HID;

  // swizzled LDS offsets of this lane's two 16B B-fragment halves
  const int swz   = ((lane >> 3) & 1) * 16;
  const int loff0 = lane * 32 + swz;
  const int loff1 = lane * 32 + (16 ^ swz);

  v8f c_reg = {};                       // cell state: resident in VGPRs
  int bstep = 0;

  #pragma unroll
  for (int L = 0; L < 2; ++L) {
    const int K     = L ? (2 * HID) : (NUMS + HID);
    const int bndry = L ? HID : NUMS;          // split point in [x_t, h]
    const int NC    = K / 32;                  // 36 or 64 K-chunks
    const __bf16* Wt  = L ? Wt1 : Wt0;
    const float* bias = L ? b1 : b0;

    // ---- stage this WG's whole weight slice into LDS (once per layer) ----
    {
      int g = threadIdx.x >> 6;                // gate
      int u = threadIdx.x & 63;
      int l = u >> 1, h = u & 1;               // target lane, 16B half
      const __bf16* src = Wt + (size_t)(g * HID + n0 + (l & 15)) * K +
                          ((l >> 4) * 8) + h * 16;
      int doff = (g << 10) + ((l * 32 + h * 16) ^ (((l >> 3) & 1) * 16));
      #pragma unroll 4
      for (int c = 0; c < NC; ++c) {
        *reinterpret_cast<uint4*>(smem + doff) =
            *reinterpret_cast<const uint4*>(src);
        src += 32;
        doff += 4096;
      }
      __syncthreads();
    }

    float bia[4];
    #pragma unroll
    for (int g = 0; g < 4; ++g) bia[g] = bias[g * HID + nlane];

    #pragma unroll 1
    for (int t = 0; t < SEQL; ++t) {
      const __bf16* xsrc = L ? (hs0 + (size_t)(t + 1) * BH)
                             : (xT + (size_t)t * BATCH * NUMS);
      const int xstride  = L ? HID : NUMS;
      const __bf16* hprev =
          L ? ((t == 0) ? (hs0 + (size_t)SEQL * BH)
                        : (h1 + (size_t)((t - 1) & 1) * BH))
            : (hs0 + (size_t)t * BH);
      __bf16* hw = L ? (h1 + (size_t)(t & 1) * BH)
                     : (hs0 + (size_t)(t + 1) * BH);

      // init accumulators with bias (C operand of first WMMA)
      v8f acc[4];
      #pragma unroll
      for (int g = 0; g < 4; ++g) {
        float bg = bia[g];
        #pragma unroll
        for (int v = 0; v < 8; ++v) acc[g][v] = bg;
      }

      int bo = 0;   // chunk-major LDS cursor, +4096/chunk; gates at +g*1024

      auto run_chunks = [&](const __bf16* ap, int chunks) {
        #pragma unroll 4
        for (int c = 0; c < chunks; ++c) {
          Frag a;
          a.q[0] = *reinterpret_cast<const uint4*>(ap);
          a.q[1] = *reinterpret_cast<const uint4*>(ap + 16);
          ap += 32;
          #pragma unroll
          for (int g = 0; g < 4; ++g) {
            Frag b;
            b.q[0] = *reinterpret_cast<const uint4*>(smem + bo + (g << 10) +
                                                     loff0);
            b.q[1] = *reinterpret_cast<const uint4*>(smem + bo + (g << 10) +
                                                     loff1);
            acc[g] = __builtin_amdgcn_wmma_f32_16x16x32_bf16(
                false, a.v, false, b.v, (short)0, acc[g], false, false);
          }
          bo += 4096;
        }
      };

      const __bf16* apx = xsrc + (size_t)(mbase + (lane & 15)) * xstride +
                          ((lane >> 4) * 8);
      const __bf16* aph = hprev + (size_t)(mbase + (lane & 15)) * HID +
                          ((lane >> 4) * 8);
      run_chunks(apx, bndry / 32);   // x_t part
      run_chunks(aph, HID / 32);     // h_{t-1} part

      const bool final_step = (L == 1) && (t == SEQL - 1);
      #pragma unroll
      for (int v = 0; v < 8; ++v) {
        int m    = mbase + v + ((lane >> 4) << 3);   // C/D layout: M=v(+8)
        float ig = fast_sigmoid(acc[0][v]);
        float fg = fast_sigmoid(acc[1][v]);
        float og = fast_sigmoid(acc[2][v]);
        float gg = fast_tanh(acc[3][v]);
        float cv = fg * c_reg[v] + ig * gg;
        c_reg[v] = cv;
        float hv = og * fast_tanh(cv);
        hw[(size_t)m * HID + nlane] = (__bf16)hv;
        if (final_step) {
          out_h[(size_t)m * HID + nlane] = hv;
          out_c[(size_t)m * HID + nlane] = cv;
        }
      }

      if (!final_step) {
        ++bstep;
        grid_barrier(bar, bstep * NWG);
        // Prefetch next step's x-side A operand (static data: xT history or
        // the fully-written layer-0 output sequence) -> global_prefetch.
        if (t + 1 < SEQL) {
          const __bf16* nx = L ? (hs0 + (size_t)(t + 2) * BH)
                               : (xT + (size_t)(t + 1) * BATCH * NUMS);
          __builtin_prefetch(nx + (size_t)threadIdx.x * 64, 0, 3);
        }
      }
    }
  }
}

// ---------------------------------------------------------------------------
// Final FC: out = tanh(h_final @ fc_w + fc_b), [128,1024]x[1024,96]
// Grid: 6 WGs (N tiles) x 256 threads (8 waves = M tiles)
// ---------------------------------------------------------------------------
__global__ void __launch_bounds__(256, 1)
fc_kernel(const __bf16* __restrict__ hfin,      // [B][H] bf16
          const __bf16* __restrict__ fcwT,      // [PRED][H] bf16
          const float* __restrict__ fcb, float* __restrict__ out) {
  const int lane  = threadIdx.x & 31;
  const int wave  = threadIdx.x >> 5;
  const int mbase = wave * 16;
  const int n0    = blockIdx.x * 16;

  v8f acc = {};
  #pragma unroll 4
  for (int kc = 0; kc < HID; kc += 32) {
    v16bf af  = load_a_frag(hfin + (size_t)mbase * HID + kc, HID, lane);
    v16bf bfv = load_b_frag(fcwT + (size_t)n0 * HID + kc, (size_t)HID, lane);
    acc = __builtin_amdgcn_wmma_f32_16x16x32_bf16(
        false, af, false, bfv, (short)0, acc, false, false);
  }
  int n    = n0 + (lane & 15);
  float bb = fcb[n];
  #pragma unroll
  for (int v = 0; v < 8; ++v) {
    int m = mbase + v + ((lane >> 4) << 3);
    out[(size_t)m * PRED + n] = fast_tanh(acc[v] + bb);
  }
}

// ---------------------------------------------------------------------------
// Launch
// ---------------------------------------------------------------------------
extern "C" void kernel_launch(void* const* d_in, const int* in_sizes, int n_in,
                              void* d_out, int out_size, void* d_ws,
                              size_t ws_size, hipStream_t stream) {
  (void)in_sizes; (void)n_in; (void)out_size; (void)ws_size;
  const float* x    = (const float*)d_in[0];   // [B][NUMS][SEQ]
  const float* W0   = (const float*)d_in[1];   // [1152][4096]
  const float* b0   = (const float*)d_in[2];   // [4096]
  const float* W1   = (const float*)d_in[3];   // [2048][4096]
  const float* b1   = (const float*)d_in[4];   // [4096]
  const float* fcw  = (const float*)d_in[5];   // [1024][96]
  const float* fcb  = (const float*)d_in[6];   // [96]

  const int K0 = NUMS + HID;                    // 1152
  const int K1 = HID + HID;                     // 2048
  const int N4 = 4 * HID;                       // 4096

  // workspace layout (all 16B aligned)
  char* w       = (char*)d_ws;
  int*  bar     = (int*)w;                               // 256 B
  __bf16* Wt0   = (__bf16*)(w + 256);                    // [4096][1152]
  __bf16* Wt1   = Wt0 + (size_t)N4 * K0;                 // [4096][2048]
  __bf16* fcwT  = Wt1 + (size_t)N4 * K1;                 // [96][1024]
  __bf16* xT    = fcwT + (size_t)PRED * HID;             // [T][B][NUMS]
  __bf16* hs0   = xT + (size_t)SEQL * BATCH * NUMS;      // [T+1][B][H]
  __bf16* h1    = hs0 + (size_t)(SEQL + 1) * BATCH * HID;// [2][B][H]

  float* out   = (float*)d_out;
  float* out_h = out + BATCH * PRED;
  float* out_c = out_h + (size_t)BATCH * HID;

  // 1) weight / input conversion (deterministic, re-done every call)
  {
    int n = K0 * N4;
    transpose_bf16<<<(n + 255) / 256, 256, 0, stream>>>(W0, Wt0, K0, N4);
  }
  {
    int n = K1 * N4;
    transpose_bf16<<<(n + 255) / 256, 256, 0, stream>>>(W1, Wt1, K1, N4);
  }
  {
    int n = HID * PRED;
    transpose_bf16<<<(n + 255) / 256, 256, 0, stream>>>(fcw, fcwT, HID, PRED);
  }
  {
    int n = SEQL * BATCH * NUMS;
    xpose_bf16<<<(n + 255) / 256, 256, 0, stream>>>(x, xT);
  }
  init_state<<<(BATCH * HID + 255) / 256, 256, 0, stream>>>(hs0, bar);

  // 2) persistent 2-layer recurrence; dynamic LDS = layer-1 slice (256KB)
  const size_t lds_bytes = (size_t)4 * (K1 / 32) * 1024;  // 262144
  lstm_persistent<<<NWG, 256, lds_bytes, stream>>>(b0, b1, Wt0, Wt1, xT, hs0,
                                                   h1, out_h, out_c, bar);

  // 3) final FC + tanh; h_final (layer1, t=255) lives at h1 + BH (255&1 == 1)
  fc_kernel<<<PRED / 16, 256, 0, stream>>>(h1 + (size_t)BATCH * HID, fcwT,
                                           fcb, out);
}
